// TriangleAttention_5806795784284
// MI455X (gfx1250) — matrix operand order repaired and checked by reading, not compiled, not run on hardware
//
#include <hip/hip_runtime.h>
#include <hip/hip_bf16.h>

// ---------------------------------------------------------------------------
// Types for CDNA5 WMMA bf16 path
// ---------------------------------------------------------------------------
typedef __attribute__((ext_vector_type(16))) __bf16 v16bf;
typedef __attribute__((ext_vector_type(8)))  __bf16 v8bf;
typedef __attribute__((ext_vector_type(8)))  float  v8f;

__device__ __forceinline__ v8f wmma_bf16(v16bf a, v16bf b, v8f c) {
  // v_wmma_f32_16x16x32_bf16 : D = A(16x32) * B(32x16) + C(16x16)
  return __builtin_amdgcn_wmma_f32_16x16x32_bf16(
      /*neg_a=*/false, a, /*neg_b=*/false, b,
      /*c_mod=*/(short)0, c, /*reuse_a=*/false, /*reuse_b=*/false);
}

__device__ __forceinline__ float fast_rcp(float x) {
  return __builtin_amdgcn_rcpf(x);   // v_rcp_f32 (sufficient for sigmoid/softmax)
}

// A-fragment (16x32 bf16). Per ISA: M = lane&15 for both halves.
// lanes 0-15 : e=0..7 -> K=k0+e         ; e=8..15 -> K=k0+16+(e-8)
// lanes 16-31: e=0..7 -> K=k0+8+e       ; e=8..15 -> K=k0+24+(e-8)
__device__ __forceinline__ v16bf load_a_frag(const __bf16* base, int row,
                                             int ld, int k0, int half) {
  const __bf16* r = base + (size_t)row * ld + k0;
  v8bf lo = *(const v8bf*)(r + half * 8);
  v8bf hi = *(const v8bf*)(r + 16 + half * 8);
  v16bf out;
#pragma unroll
  for (int e = 0; e < 8; ++e) { out[e] = lo[e]; out[e + 8] = hi[e]; }
  return out;
}

// Same A layout but source is an f32 LDS buffer (softmax probabilities).
__device__ __forceinline__ v16bf load_a_frag_f32(const float* base, int row,
                                                 int ld, int k0, int half) {
  const float* r = base + (size_t)row * ld + k0;
  v16bf out;
#pragma unroll
  for (int e = 0; e < 8; ++e) {
    out[e]     = (__bf16)r[half * 8 + e];
    out[e + 8] = (__bf16)r[16 + half * 8 + e];
  }
  return out;
}

// B-fragment (32x16 bf16) from a column-major-transposed matrix wT[col][k]:
// N = lane&15 (column), e=0..7 -> K=k0+half*8+e ; e=8..15 -> K=k0+16+half*8+(e-8).
__device__ __forceinline__ v16bf load_b_frag(const __bf16* baseT, int col,
                                             int ld, int k0, int half) {
  const __bf16* r = baseT + (size_t)col * ld + k0;
  v8bf lo = *(const v8bf*)(r + half * 8);
  v8bf hi = *(const v8bf*)(r + 16 + half * 8);
  v16bf out;
#pragma unroll
  for (int e = 0; e < 8; ++e) { out[e] = lo[e]; out[e + 8] = hi[e]; }
  return out;
}

// C/D element (reg rg): row = half*8 + rg, col = lane&15 (within 16x16 tile).

#define NSEQ   256
#define INDIM  128
#define NH     4
#define HD     32

// ---------------------------------------------------------------------------
// Kernel 0: convert + transpose weights to bf16 [col][k] layout.
//   wT1  : [512][128]  cols 0..383 = w_qkv, 384..511 = w_gate
//   woutT: [128][128]
// ---------------------------------------------------------------------------
__global__ void k0_wconv(const float* __restrict__ w_qkv,
                         const float* __restrict__ w_gate,
                         const float* __restrict__ w_out,
                         __bf16* __restrict__ wT1,
                         __bf16* __restrict__ woutT) {
  int idx = blockIdx.x * 256 + threadIdx.x;
  if (idx < 512 * 128) {
    int col = idx >> 7, k = idx & 127;
    float v = (col < 384) ? w_qkv[k * 384 + col] : w_gate[k * 128 + (col - 384)];
    wT1[idx] = (__bf16)v;
  }
  int idx2 = idx - 512 * 128;
  if (idx2 >= 0 && idx2 < 128 * 128) {
    int col = idx2 >> 7, k = idx2 & 127;
    woutT[idx2] = (__bf16)w_out[k * 128 + col];
  }
}

// ---------------------------------------------------------------------------
// Kernel 1: fused LayerNorm + {qkv, pair-bias, gate} projections.
// Block: 256 threads (8 waves), 64 pixels. Grid: 65536/64 = 1024 blocks.
//   q/k/v_ws : bf16 [r][h][i][d]
//   pb_ws    : f32  [h][i][j]
//   gate_ws  : f32  [p][128]
// ---------------------------------------------------------------------------
__global__ __launch_bounds__(256) void k1_ln_proj(
    const float* __restrict__ z, const float* __restrict__ ln_g,
    const float* __restrict__ ln_b, const float* __restrict__ w_pair,
    const float* __restrict__ b_gate, const __bf16* __restrict__ wT1,
    __bf16* __restrict__ q_ws, __bf16* __restrict__ k_ws,
    __bf16* __restrict__ v_ws, float* __restrict__ pb_ws,
    float* __restrict__ gate_ws) {
  __shared__ __bf16 zn[64][INDIM];
  __shared__ float red_s[64][4];
  __shared__ float red_q[64][4];

  int t = threadIdx.x;
  int pbase = blockIdx.x * 64;

  // ---- LayerNorm: 4 threads per pixel, each owns 32 channels ----
  {
    int pl = t >> 2, qd = t & 3;
    const float* row = z + (size_t)(pbase + pl) * INDIM + qd * 32;
    float vals[32];
    float s = 0.f, sq = 0.f;
#pragma unroll
    for (int c = 0; c < 32; ++c) {
      float x = row[c];
      vals[c] = x; s += x; sq += x * x;
    }
    red_s[pl][qd] = s; red_q[pl][qd] = sq;
    __syncthreads();
    float ts = red_s[pl][0] + red_s[pl][1] + red_s[pl][2] + red_s[pl][3];
    float tq = red_q[pl][0] + red_q[pl][1] + red_q[pl][2] + red_q[pl][3];
    float mu  = ts * (1.f / 128.f);
    float var = tq * (1.f / 128.f) - mu * mu;
    float inv = rsqrtf(var + 1e-5f);
#pragma unroll
    for (int c = 0; c < 32; ++c) {
      int col = qd * 32 + c;
      float zv = (vals[c] - mu) * inv * ln_g[col] + ln_b[col];
      zn[pl][col] = (__bf16)zv;
    }
  }
  __syncthreads();

  // ---- pair bias: thread -> (pixel t/4, head t%4), f32 dot of length 128 ----
  {
    int pl = t >> 2, h = t & 3;
    float acc = 0.f;
    for (int c = 0; c < INDIM; ++c)
      acc += (float)zn[pl][c] * w_pair[c * NH + h];
    int p = pbase + pl;
    int i = p >> 8, j = p & 255;
    pb_ws[((size_t)(h * NSEQ + i)) * NSEQ + j] = acc;
  }

  // ---- GEMM: zn(64x128) @ [wqkv|wgate](128x512) via WMMA ----
  int wv = t >> 5, lane = t & 31, half = lane >> 4, l15 = lane & 15;
  int ri = pbase >> 8;              // block-uniform: row index of pixel grid
  int iibase = pbase & 255;         // block-uniform: first in-row pixel

  for (int mt = 0; mt < 4; ++mt) {
    v16bf afr[4];
#pragma unroll
    for (int kk = 0; kk < 4; ++kk)
      afr[kk] = load_a_frag(&zn[0][0], mt * 16 + l15, INDIM, kk * 32, half);

    // q/k/v tiles: nt in {wv, wv+8, wv+16}, all < 24 -> no divergence at all
    for (int nt = wv; nt < 24; nt += 8) {
      int cb = nt * 16;
      v16bf bfr[4];
#pragma unroll
      for (int kk = 0; kk < 4; ++kk)
        bfr[kk] = load_b_frag(wT1, cb + l15, INDIM, kk * 32, half);
      v8f acc = {0.f, 0.f, 0.f, 0.f, 0.f, 0.f, 0.f, 0.f};
#pragma unroll
      for (int kk = 0; kk < 4; ++kk)
        acc = wmma_bf16(afr[kk], bfr[kk], acc);

      int which = cb >> 7;                  // 0=q 1=k 2=v   (tile-uniform)
      int h     = (cb & 127) >> 5;          // head          (tile-uniform)
      int d     = (cb & 31) + l15;          // lane-varying only
      __bf16* dst = (which == 0) ? q_ws : (which == 1) ? k_ws : v_ws;
      int ii0 = iibase + mt * 16 + half * 8;
      size_t tb = ((size_t)(ri * NH + h) * NSEQ + ii0) * HD + d;
#pragma unroll
      for (int rg = 0; rg < 8; ++rg)
        dst[tb + (size_t)rg * HD] = (__bf16)acc[rg];
    }

    // gate tile: exactly one per wave, nt = 24 + wv
    {
      int cb = (24 + wv) * 16;
      v16bf bfr[4];
#pragma unroll
      for (int kk = 0; kk < 4; ++kk)
        bfr[kk] = load_b_frag(wT1, cb + l15, INDIM, kk * 32, half);
      v8f acc = {0.f, 0.f, 0.f, 0.f, 0.f, 0.f, 0.f, 0.f};
#pragma unroll
      for (int kk = 0; kk < 4; ++kk)
        acc = wmma_bf16(afr[kk], bfr[kk], acc);

      int gc = (cb - 384) + l15;
      float bg = b_gate[gc];
      size_t gb = (size_t)(pbase + mt * 16 + half * 8) * INDIM + gc;
#pragma unroll
      for (int rg = 0; rg < 8; ++rg) {
        float xv = acc[rg] + bg;
        float g = fast_rcp(1.f + __expf(-xv));   // sigmoid via v_rcp_f32
        gate_ws[gb + (size_t)rg * INDIM] = g;
      }
    }
  }
}

// ---------------------------------------------------------------------------
// Kernel 2: attention for one (r, h, 32-row i-block). Block: 64 threads
// (2 waves, one 16-row i-tile each). QK^T WMMA (K=D=32, single instr/tile),
// scale + pair bias + faithful -1e-9 masking, LDS softmax over j=256,
// attn@V WMMA from LDS-transposed V, gated bf16 output.
// ---------------------------------------------------------------------------
__global__ __launch_bounds__(64) void k2_attn(
    const __bf16* __restrict__ q_ws, const __bf16* __restrict__ k_ws,
    const __bf16* __restrict__ v_ws, const float* __restrict__ pb_ws,
    const float* __restrict__ gate_ws, const int* __restrict__ src_mask,
    __bf16* __restrict__ x_ws) {
  __shared__ float  scores[2][16][NSEQ];   // 32 KB
  __shared__ __bf16 vT[HD][NSEQ];          // 16 KB
  __shared__ float  mf[NSEQ];              //  1 KB
  __shared__ float  redmax[2][2][16];
  __shared__ float  redsum[2][2][16];

  int r = blockIdx.z, h = blockIdx.y, ib = blockIdx.x;
  int t = threadIdx.x;
  size_t base = ((size_t)(r * NH + h)) * NSEQ * HD;
  const __bf16* qp = q_ws + base;
  const __bf16* kp = k_ws + base;
  const __bf16* vp = v_ws + base;

  // stage mask sign values and transposed V
  for (int j = t; j < NSEQ; j += 64) mf[j] = src_mask[j] ? 1.f : -1.f;
  for (int idx = t; idx < NSEQ * HD; idx += 64) {
    int j = idx >> 5, d = idx & 31;
    vT[d][j] = vp[j * HD + d];
  }
  __syncthreads();

  int wv = t >> 5, lane = t & 31, half = lane >> 4, l15 = lane & 15;
  int i0 = ib * 32 + wv * 16;
  const float scale = 0.17677669529663687f;  // 1/sqrt(32)

  v16bf qfr = load_a_frag(qp, i0 + l15, HD, 0, half);
  const float* pbrow = pb_ws + ((size_t)(h * NSEQ + i0)) * NSEQ;

  // ---- logits = q.k*scale + pb, with mask quirk ----
  for (int jt = 0; jt < 16; ++jt) {
    int j0 = jt * 16;
    v16bf kfr = load_b_frag(kp, j0 + l15, HD, 0, half);
    v8f acc = {0.f, 0.f, 0.f, 0.f, 0.f, 0.f, 0.f, 0.f};
    acc = wmma_bf16(qfr, kfr, acc);
    int j = j0 + l15;
    float mj = mf[j];
#pragma unroll
    for (int rg = 0; rg < 8; ++rg) {
      int m = half * 8 + rg;
      float val = acc[rg] * scale + pbrow[(size_t)m * NSEQ + j];
      if (mf[i0 + m] * mj == -1.f) val = -1e-9f;  // faithful to reference
      scores[wv][m][j] = val;
    }
  }
  __syncthreads();

  // ---- softmax over j: 2 lanes per row (half splits the 256 columns).
  //      Pass B caches exp() in-place so normalization is a pure multiply. ----
  {
    int row = l15;
    float* sr = scores[wv][row];
    float mx = -3.4e38f;
    for (int j = half * 128; j < half * 128 + 128; ++j) mx = fmaxf(mx, sr[j]);
    redmax[wv][half][row] = mx;
    __syncthreads();
    float rmx = fmaxf(redmax[wv][0][row], redmax[wv][1][row]);
    float sm = 0.f;
    for (int j = half * 128; j < half * 128 + 128; ++j) {
      float e = __expf(sr[j] - rmx);
      sr[j] = e;                 // cache exp value
      sm += e;
    }
    redsum[wv][half][row] = sm;
    __syncthreads();
    float inv = fast_rcp(redsum[wv][0][row] + redsum[wv][1][row]);
    for (int j = half * 128; j < half * 128 + 128; ++j)
      sr[j] = sr[j] * inv;       // pure multiply, no second exp
  }
  __syncthreads();

  // ---- out = attn @ V (two 16x16 d-tiles, K loop over j in chunks of 32) ----
  v8f oacc0 = {0.f, 0.f, 0.f, 0.f, 0.f, 0.f, 0.f, 0.f};
  v8f oacc1 = {0.f, 0.f, 0.f, 0.f, 0.f, 0.f, 0.f, 0.f};
  for (int jb = 0; jb < 8; ++jb) {
    v16bf afr = load_a_frag_f32(&scores[wv][0][0], l15, NSEQ, jb * 32, half);
    v16bf b0 = load_b_frag(&vT[0][0], 0 + l15, NSEQ, jb * 32, half);
    v16bf b1 = load_b_frag(&vT[0][0], 16 + l15, NSEQ, jb * 32, half);
    oacc0 = wmma_bf16(afr, b0, oacc0);
    oacc1 = wmma_bf16(afr, b1, oacc1);
  }

  // ---- epilogue: apply sigmoid gate, store bf16 x = gate * attn_out ----
  {
    size_t p0 = (size_t)r * NSEQ + i0 + half * 8;
    int c0 = h * HD + l15;        // d-tile 0
    int c1 = h * HD + 16 + l15;   // d-tile 1
#pragma unroll
    for (int rg = 0; rg < 8; ++rg) {
      size_t p = p0 + rg;
      x_ws[p * INDIM + c0] = (__bf16)(gate_ws[p * INDIM + c0] * oacc0[rg]);
      x_ws[p * INDIM + c1] = (__bf16)(gate_ws[p * INDIM + c1] * oacc1[rg]);
    }
  }
}

// ---------------------------------------------------------------------------
// Kernel 3: final = x @ w_out + b_out.  Block: 256 threads (8 waves),
// 128 rows/block, each wave owns one 16-row M-tile x 8 N-tiles.
// ---------------------------------------------------------------------------
__global__ __launch_bounds__(256) void k3_out(
    const __bf16* __restrict__ x_ws, const __bf16* __restrict__ woutT,
    const float* __restrict__ b_out, float* __restrict__ out) {
  int t = threadIdx.x;
  int wv = t >> 5, lane = t & 31, half = lane >> 4, l15 = lane & 15;
  int mrow = blockIdx.x * 128 + wv * 16;

  v16bf afr[4];
#pragma unroll
  for (int kk = 0; kk < 4; ++kk)
    afr[kk] = load_a_frag(x_ws, mrow + l15, INDIM, kk * 32, half);

  for (int nt = 0; nt < 8; ++nt) {
    int cb = nt * 16;
    v16bf bfr[4];
#pragma unroll
    for (int kk = 0; kk < 4; ++kk)
      bfr[kk] = load_b_frag(woutT, cb + l15, INDIM, kk * 32, half);
    v8f acc = {0.f, 0.f, 0.f, 0.f, 0.f, 0.f, 0.f, 0.f};
#pragma unroll
    for (int kk = 0; kk < 4; ++kk)
      acc = wmma_bf16(afr[kk], bfr[kk], acc);
    int col = cb + l15;
    float bo = b_out[col];
#pragma unroll
    for (int rg = 0; rg < 8; ++rg) {
      int m = half * 8 + rg;
      out[(size_t)(mrow + m) * INDIM + col] = acc[rg] + bo;
    }
  }
}

// ---------------------------------------------------------------------------
// Launch
// ---------------------------------------------------------------------------
extern "C" void kernel_launch(void* const* d_in, const int* in_sizes, int n_in,
                              void* d_out, int out_size, void* d_ws,
                              size_t ws_size, hipStream_t stream) {
  const float* z        = (const float*)d_in[0];
  const int*   src_mask = (const int*)d_in[1];
  const float* ln_g     = (const float*)d_in[2];
  const float* ln_b     = (const float*)d_in[3];
  const float* w_qkv    = (const float*)d_in[4];
  const float* w_pair   = (const float*)d_in[5];
  const float* w_gate   = (const float*)d_in[6];
  const float* b_gate   = (const float*)d_in[7];
  const float* w_out    = (const float*)d_in[8];
  const float* b_out    = (const float*)d_in[9];
  float* out = (float*)d_out;

  char* ws = (char*)d_ws;
  const size_t P    = (size_t)NSEQ * NSEQ;      // 65536 pixels
  const size_t QKV  = P * INDIM;                // 8,388,608 elements
  __bf16* wT1    = (__bf16*)ws;                            // 512*128 bf16
  __bf16* woutT  = (__bf16*)(ws + 131072);                 // 128*128 bf16
  __bf16* q_ws   = (__bf16*)(ws + 163840);
  __bf16* k_ws   = q_ws + QKV;
  __bf16* v_ws   = k_ws + QKV;
  float*  pb_ws  = (float*)(ws + 163840 + 3 * QKV * sizeof(__bf16));
  float*  gate_ws = pb_ws + (size_t)NH * NSEQ * NSEQ;
  __bf16* x_ws   = (__bf16*)((char*)gate_ws + QKV * sizeof(float));

  k0_wconv<<<320, 256, 0, stream>>>(w_qkv, w_gate, w_out, wT1, woutT);
  k1_ln_proj<<<(int)(P / 64), 256, 0, stream>>>(
      z, ln_g, ln_b, w_pair, b_gate, wT1, q_ws, k_ws, v_ws, pb_ws, gate_ws);
  dim3 g2(8, NH, NSEQ);
  k2_attn<<<g2, 64, 0, stream>>>(q_ws, k_ws, v_ws, pb_ws, gate_ws, src_mask,
                                 x_ws);
  k3_out<<<(int)(P / 128), 256, 0, stream>>>(x_ws, woutT, b_out, out);
}
